// EdgeConvBlock_88811333747380
// MI455X (gfx1250) — matrix operand order repaired and checked by reading, not compiled, not run on hardware
//
#include <hip/hip_runtime.h>

// EdgeConv block for MI455X (gfx1250, wave32, WMMA bf16 16x16x32, f32 accum).
// All GEMMs use v_wmma_f32_16x16x32_bf16. Top-K fused into the distance GEMM
// using the 320KB WGP LDS (16 x 4096 f32 distance tile resident in LDS).
// BN statistics are computed via deterministic per-WG partials (no fp atomics).
// global_prefetch_b8 used to hide gather latency in the edge-MLP kernels.

#define BB    8
#define NN    4096
#define KNB   20
#define CIN   64
#define CH    64
#define ROWS_TOTAL   (BB * NN)            // 32768 points
#define EROWS        (ROWS_TOTAL * KNB)   // 655360 edge rows
#define ETILES       (EROWS / 16)         // 40960 16-row tiles

typedef __attribute__((ext_vector_type(16))) __bf16 v16bf;
typedef __attribute__((ext_vector_type(8)))  __bf16 v8bf;
typedef __attribute__((ext_vector_type(8)))  float  v8f;

static __device__ __forceinline__ __bf16 f2bf(float f) {
  unsigned u = __builtin_bit_cast(unsigned, f);
  u += 0x7FFFu + ((u >> 16) & 1u);                  // round-to-nearest-even
  unsigned short s = (unsigned short)(u >> 16);
  return __builtin_bit_cast(__bf16, s);
}

struct f8 { float4 a, b; };

static __device__ __forceinline__ f8 load8(const float* p) {
  f8 r; r.a = *(const float4*)p; r.b = *(const float4*)(p + 4); return r;
}
static __device__ __forceinline__ f8 sub8(const f8& p, const f8& q) {
  f8 r;
  r.a.x = p.a.x - q.a.x; r.a.y = p.a.y - q.a.y; r.a.z = p.a.z - q.a.z; r.a.w = p.a.w - q.a.w;
  r.b.x = p.b.x - q.b.x; r.b.y = p.b.y - q.b.y; r.b.z = p.b.z - q.b.z; r.b.w = p.b.w - q.b.w;
  return r;
}
// h*a + c then ReLU, componentwise (fused BN-apply)
static __device__ __forceinline__ f8 bnrelu8(const f8& h, const f8& a, const f8& c) {
  f8 r;
  r.a.x = fmaxf(fmaf(h.a.x, a.a.x, c.a.x), 0.f);
  r.a.y = fmaxf(fmaf(h.a.y, a.a.y, c.a.y), 0.f);
  r.a.z = fmaxf(fmaf(h.a.z, a.a.z, c.a.z), 0.f);
  r.a.w = fmaxf(fmaf(h.a.w, a.a.w, c.a.w), 0.f);
  r.b.x = fmaxf(fmaf(h.b.x, a.b.x, c.b.x), 0.f);
  r.b.y = fmaxf(fmaf(h.b.y, a.b.y, c.b.y), 0.f);
  r.b.z = fmaxf(fmaf(h.b.z, a.b.z, c.b.z), 0.f);
  r.b.w = fmaxf(fmaf(h.b.w, a.b.w, c.b.w), 0.f);
  return r;
}
static __device__ __forceinline__ void pack8(v16bf& d, int base, const f8& v) {
  d[base + 0] = f2bf(v.a.x); d[base + 1] = f2bf(v.a.y);
  d[base + 2] = f2bf(v.a.z); d[base + 3] = f2bf(v.a.w);
  d[base + 4] = f2bf(v.b.x); d[base + 5] = f2bf(v.b.y);
  d[base + 6] = f2bf(v.b.z); d[base + 7] = f2bf(v.b.w);
}
static __device__ __forceinline__ v16bf mk16(v8bf lo, v8bf hi) {
  v16bf r;
#pragma unroll
  for (int i = 0; i < 8; ++i) { r[i] = lo[i]; r[i + 8] = hi[i]; }
  return r;
}
#define WMMA_BF16(A, B, C) \
  __builtin_amdgcn_wmma_f32_16x16x32_bf16(false, (A), false, (B), (short)0, (C), false, false)

// ---------------------------------------------------------------------------
// Kernel: x -> bf16 copy + squared norms. One wave per point row (64 ch).
// ---------------------------------------------------------------------------
__global__ void k_prep_x(const float* __restrict__ x, __bf16* __restrict__ xbf,
                         float* __restrict__ sq) {
  int lane = threadIdx.x & 31;
  int row  = blockIdx.x * 8 + (threadIdx.x >> 5);
  const float* xr = x + (size_t)row * CIN;
  float v0 = xr[lane], v1 = xr[lane + 32];
  xbf[(size_t)row * CIN + lane]      = f2bf(v0);
  xbf[(size_t)row * CIN + lane + 32] = f2bf(v1);
  float s = v0 * v0 + v1 * v1;
#pragma unroll
  for (int off = 16; off > 0; off >>= 1) s += __shfl_xor(s, off, 32);
  if (lane == 0) sq[row] = s;
}

// ---------------------------------------------------------------------------
// Kernel: transpose + convert W1 [128,64] -> W1t [64][128] bf16,
//                              W2 [64,64]  -> W2t [64][64]  bf16.
// ---------------------------------------------------------------------------
__global__ void k_prep_w(const float* __restrict__ W1, const float* __restrict__ W2,
                         __bf16* __restrict__ W1t, __bf16* __restrict__ W2t) {
  int idx = blockIdx.x * 256 + threadIdx.x;
  if (idx < 2 * CIN * CH) {                       // 8192
    int k = idx / CH, n = idx - k * CH;
    W1t[n * (2 * CIN) + k] = f2bf(W1[idx]);
  } else if (idx < 2 * CIN * CH + CH * CH) {      // +4096
    int e = idx - 2 * CIN * CH;
    int k = e / CH, n = e - k * CH;
    W2t[n * CH + k] = f2bf(W2[e]);
  }
}

// ---------------------------------------------------------------------------
// Kernel: fused distance GEMM (bf16 WMMA) + top-20 selection in LDS.
// 1 WG (8 waves, 256 thr) = 16 rows of one batch; 2 waves per SIMD32.
// LDS tile: 16 x 4100 f32 (~256KB, row stride padded to avoid bank conflicts).
// ---------------------------------------------------------------------------
__global__ void k_knn(const __bf16* __restrict__ xbf, const float* __restrict__ sq,
                      int* __restrict__ knn) {
  extern __shared__ float lds[];
  const int RS = 4100;
  int wg   = blockIdx.x;
  int b    = wg >> 8;                 // NN/16 = 256 tiles per batch
  int row0 = (wg & 255) << 4;
  int tid = threadIdx.x, lane = tid & 31, wid = tid >> 5;   // 8 waves
  const __bf16* xb = xbf + (size_t)b * NN * CIN;
  const float*  sb = sq + b * NN;
  int m = lane & 15, half = lane >> 4, c0 = half * 8;

  // A fragments: rows row0..row0+15, K = 0..63 (two 16x32 bf16 fragments)
  const __bf16* ar = xb + (size_t)(row0 + m) * CIN;
  v16bf a0 = mk16(*(const v8bf*)(ar + c0),      *(const v8bf*)(ar + c0 + 16));
  v16bf a1 = mk16(*(const v8bf*)(ar + 32 + c0), *(const v8bf*)(ar + 32 + c0 + 16));

  float rsq[8];
#pragma unroll
  for (int r = 0; r < 8; ++r) rsq[r] = sb[row0 + r + 8 * half];

  // Sweep all 4096 columns: 8 waves x 16 cols per iteration, 32 iterations.
  int n = lane & 15, kh = lane >> 4;
  for (int it = 0; it < 32; ++it) {
    int col0 = ((it << 3) + wid) << 4;
    const __bf16* br = xb + (size_t)(col0 + n) * CIN + kh * 8;   // rows of X as B cols
    // prefetch next column block (this wave's next 16 rows of X)
    if (it < 31)
      __builtin_prefetch(br + 128 * CIN, 0, 1);
    v16bf b0 = mk16(*(const v8bf*)(br),      *(const v8bf*)(br + 16));
    v16bf b1 = mk16(*(const v8bf*)(br + 32), *(const v8bf*)(br + 48));
    v8f acc = {};
    acc = WMMA_BF16(a0, b0, acc);
    acc = WMMA_BF16(a1, b1, acc);
    float cs = sb[col0 + n];
#pragma unroll
    for (int r = 0; r < 8; ++r) {
      float d2 = rsq[r] + cs - 2.0f * acc[r];
      lds[(r + 8 * half) * RS + col0 + n] = d2;
    }
  }
  __syncthreads();

  // Selection: each wave owns 2 rows; 20x wave-wide argmin over 4096 values.
  for (int rr = 0; rr < 2; ++rr) {
    int m2 = (wid << 1) + rr;
    float* dr = lds + m2 * RS;
    int* orow = knn + ((size_t)(b * NN + row0 + m2)) * KNB;
    for (int t = 0; t < KNB; ++t) {
      float best = 3.402823e38f; int bi = NN;
      for (int j = lane; j < NN; j += 32) {
        float v = dr[j];
        if (v < best) { best = v; bi = j; }
      }
#pragma unroll
      for (int off = 16; off > 0; off >>= 1) {
        float ov = __shfl_xor(best, off, 32);
        int   oi = __shfl_xor(bi,   off, 32);
        if (ov < best || (ov == best && oi < bi)) { best = ov; bi = oi; }
      }
      if (lane == 0) { orow[t] = bi; dr[bi] = 3.402823e38f; }  // pop winner
    }
  }
}

// ---------------------------------------------------------------------------
// Kernel: edge-feature gather + layer-1 GEMM.
// A row = [ctr(64) | nbr-ctr(64)] built on the fly (bf16), B = W1t.
// Epilogue: +b1, store h1 (f32), accumulate per-channel sum/sumsq partials.
// ---------------------------------------------------------------------------
__global__ void k_gemm1(const float* __restrict__ x, const int* __restrict__ knn,
                        const __bf16* __restrict__ W1t, const float* __restrict__ b1,
                        float* __restrict__ h1, float* __restrict__ parts) {
  __shared__ float s_red[4][2][64];
  int tid = threadIdx.x, lane = tid & 31, wid = tid >> 5;
  int gw = blockIdx.x * 4 + wid;                  // 4096 waves total
  int m = lane & 15, half = lane >> 4, c0 = half * 8;
  float psum[4] = {0, 0, 0, 0}, psq[4] = {0, 0, 0, 0};

  for (int tile = gw; tile < ETILES; tile += 4096) {
    int row0 = tile << 4;
    int rm   = row0 + m;
    int node = rm / KNB;
    int kk   = rm - node * KNB;
    int bb   = node >> 12;                        // node / NN
    int j    = knn[node * KNB + kk];
    const float* ctr = x + (size_t)node * CIN;
    const float* nbr = x + (size_t)((bb << 12) + j) * CIN;

    // prefetch next tile's gather chain (index row + center row)
    int ntile = tile + 4096;
    if (ntile < ETILES) {
      int nrm = (ntile << 4) + m;
      int nnode = nrm / KNB;
      __builtin_prefetch(knn + nnode * KNB + (nrm - nnode * KNB), 0, 1);
      __builtin_prefetch(x + (size_t)nnode * CIN, 0, 1);
    }

    f8 cA0 = load8(ctr + c0),      cA1 = load8(ctr + c0 + 16);
    f8 cB0 = load8(ctr + 32 + c0), cB1 = load8(ctr + 32 + c0 + 16);
    f8 nA0 = load8(nbr + c0),      nA1 = load8(nbr + c0 + 16);
    f8 nB0 = load8(nbr + 32 + c0), nB1 = load8(nbr + 32 + c0 + 16);

    v16bf a0, a1, a2, a3;
    pack8(a0, 0, cA0); pack8(a0, 8, cA1);                 // K   0..31  : ctr
    pack8(a1, 0, cB0); pack8(a1, 8, cB1);                 // K  32..63  : ctr
    f8 dA0 = sub8(nA0, cA0), dA1 = sub8(nA1, cA1);
    f8 dB0 = sub8(nB0, cB0), dB1 = sub8(nB1, cB1);
    pack8(a2, 0, dA0); pack8(a2, 8, dA1);                 // K  64..95  : nbr-ctr
    pack8(a3, 0, dB0); pack8(a3, 8, dB1);                 // K  96..127 : nbr-ctr

    int n = lane & 15, kh = half;
    for (int colt = 0; colt < 4; ++colt) {
      const __bf16* wp = W1t + (size_t)(colt * 16 + n) * 128 + kh * 8;
      v16bf w0 = mk16(*(const v8bf*)(wp),       *(const v8bf*)(wp + 16));
      v16bf w1 = mk16(*(const v8bf*)(wp + 32),  *(const v8bf*)(wp + 48));
      v16bf w2 = mk16(*(const v8bf*)(wp + 64),  *(const v8bf*)(wp + 80));
      v16bf w3 = mk16(*(const v8bf*)(wp + 96),  *(const v8bf*)(wp + 112));
      v8f acc = {};
      acc = WMMA_BF16(a0, w0, acc);
      acc = WMMA_BF16(a1, w1, acc);
      acc = WMMA_BF16(a2, w2, acc);
      acc = WMMA_BF16(a3, w3, acc);
      int col = colt * 16 + n;
      float bias = b1[col];
#pragma unroll
      for (int r = 0; r < 8; ++r) {
        float h = acc[r] + bias;
        int orow = row0 + r + 8 * half;           // C-layout rows
        h1[(size_t)orow * CH + col] = h;
        psum[colt] += h; psq[colt] += h * h;
      }
    }
  }
  // stats: combine lane pairs (same output col), then cross-wave in LDS
#pragma unroll
  for (int colt = 0; colt < 4; ++colt) {
    psum[colt] += __shfl_xor(psum[colt], 16, 32);
    psq[colt]  += __shfl_xor(psq[colt], 16, 32);
  }
  if (lane < 16) {
#pragma unroll
    for (int colt = 0; colt < 4; ++colt) {
      s_red[wid][0][colt * 16 + lane] = psum[colt];
      s_red[wid][1][colt * 16 + lane] = psq[colt];
    }
  }
  __syncthreads();
  if (tid < 128) {
    int stat = tid >> 6, ch = tid & 63;
    float t = s_red[0][stat][ch] + s_red[1][stat][ch] +
              s_red[2][stat][ch] + s_red[3][stat][ch];
    parts[(size_t)blockIdx.x * 128 + stat * 64 + ch] = t;  // deterministic partials
  }
}

// ---------------------------------------------------------------------------
// Kernel: layer-2 GEMM, BN1+ReLU fused into A construction, IN-PLACE on h.
// ---------------------------------------------------------------------------
__global__ void k_gemm2(float* __restrict__ h, const __bf16* __restrict__ W2t,
                        const float* __restrict__ b2, const float* __restrict__ a1c,
                        const float* __restrict__ c1c, float* __restrict__ parts) {
  __shared__ float s_red[4][2][64];
  int tid = threadIdx.x, lane = tid & 31, wid = tid >> 5;
  int gw = blockIdx.x * 4 + wid;
  int m = lane & 15, half = lane >> 4, c0 = half * 8;
  float psum[4] = {0, 0, 0, 0}, psq[4] = {0, 0, 0, 0};

  f8 sA0 = load8(a1c + c0),      sA1 = load8(a1c + c0 + 16);
  f8 sB0 = load8(a1c + 32 + c0), sB1 = load8(a1c + 32 + c0 + 16);
  f8 tA0 = load8(c1c + c0),      tA1 = load8(c1c + c0 + 16);
  f8 tB0 = load8(c1c + 32 + c0), tB1 = load8(c1c + 32 + c0 + 16);

  for (int tile = gw; tile < ETILES; tile += 4096) {
    int row0 = tile << 4;
    const float* hr = h + (size_t)(row0 + m) * CH;
    int ntile = tile + 4096;
    if (ntile < ETILES)
      __builtin_prefetch(h + (size_t)((ntile << 4) + m) * CH, 0, 1);

    f8 hA0 = load8(hr + c0),      hA1 = load8(hr + c0 + 16);
    f8 hB0 = load8(hr + 32 + c0), hB1 = load8(hr + 32 + c0 + 16);

    v16bf a0, a1;
    pack8(a0, 0, bnrelu8(hA0, sA0, tA0)); pack8(a0, 8, bnrelu8(hA1, sA1, tA1));
    pack8(a1, 0, bnrelu8(hB0, sB0, tB0)); pack8(a1, 8, bnrelu8(hB1, sB1, tB1));

    int n = lane & 15, kh = half;
    for (int colt = 0; colt < 4; ++colt) {
      const __bf16* wp = W2t + (size_t)(colt * 16 + n) * 64 + kh * 8;
      v16bf w0 = mk16(*(const v8bf*)(wp),      *(const v8bf*)(wp + 16));
      v16bf w1 = mk16(*(const v8bf*)(wp + 32), *(const v8bf*)(wp + 48));
      v8f acc = {};
      acc = WMMA_BF16(a0, w0, acc);
      acc = WMMA_BF16(a1, w1, acc);
      int col = colt * 16 + n;
      float bias = b2[col];
#pragma unroll
      for (int r = 0; r < 8; ++r) {
        float v = acc[r] + bias;
        int orow = row0 + r + 8 * half;
        h[(size_t)orow * CH + col] = v;           // in-place (rows read above)
        psum[colt] += v; psq[colt] += v * v;
      }
    }
  }
#pragma unroll
  for (int colt = 0; colt < 4; ++colt) {
    psum[colt] += __shfl_xor(psum[colt], 16, 32);
    psq[colt]  += __shfl_xor(psq[colt], 16, 32);
  }
  if (lane < 16) {
#pragma unroll
    for (int colt = 0; colt < 4; ++colt) {
      s_red[wid][0][colt * 16 + lane] = psum[colt];
      s_red[wid][1][colt * 16 + lane] = psq[colt];
    }
  }
  __syncthreads();
  if (tid < 128) {
    int stat = tid >> 6, ch = tid & 63;
    float t = s_red[0][stat][ch] + s_red[1][stat][ch] +
              s_red[2][stat][ch] + s_red[3][stat][ch];
    parts[(size_t)blockIdx.x * 128 + stat * 64 + ch] = t;
  }
}

// ---------------------------------------------------------------------------
// Kernel: BN2+ReLU then max over K neighbors; stats for BN3. 1 wave = 1 node.
// ---------------------------------------------------------------------------
__global__ void k_maxk(const float* __restrict__ h2, const float* __restrict__ a2c,
                       const float* __restrict__ c2c, float* __restrict__ hmax,
                       float* __restrict__ parts) {
  __shared__ float s_red[8][2][64];
  int tid = threadIdx.x, lane = tid & 31, wid = tid >> 5;
  int gw = blockIdx.x * 8 + wid;                  // 1024 waves total
  int ch0 = lane, ch1 = lane + 32;
  float A0 = a2c[ch0], C0 = c2c[ch0], A1 = a2c[ch1], C1 = c2c[ch1];
  float s0 = 0, q0 = 0, s1 = 0, q1 = 0;

  for (int node = gw; node < ROWS_TOTAL; node += 1024) {
    const float* base = h2 + (size_t)node * KNB * CH;
    float m0 = -3.402823e38f, m1 = -3.402823e38f;
#pragma unroll
    for (int k = 0; k < KNB; ++k) {
      m0 = fmaxf(m0, fmaxf(fmaf(base[k * CH + ch0], A0, C0), 0.f));
      m1 = fmaxf(m1, fmaxf(fmaf(base[k * CH + ch1], A1, C1), 0.f));
    }
    hmax[(size_t)node * CH + ch0] = m0;
    hmax[(size_t)node * CH + ch1] = m1;
    s0 += m0; q0 += m0 * m0; s1 += m1; q1 += m1 * m1;
  }
  s_red[wid][0][ch0] = s0; s_red[wid][0][ch1] = s1;
  s_red[wid][1][ch0] = q0; s_red[wid][1][ch1] = q1;
  __syncthreads();
  if (tid < 128) {
    int stat = tid >> 6, ch = tid & 63;
    float t = 0;
#pragma unroll
    for (int w = 0; w < 8; ++w) t += s_red[w][stat][ch];
    parts[(size_t)blockIdx.x * 128 + stat * 64 + ch] = t;
  }
}

// ---------------------------------------------------------------------------
// Kernel: reduce partials -> fold BN into per-channel scale a, shift c.
// ---------------------------------------------------------------------------
__global__ void k_fin(const float* __restrict__ parts, int nparts,
                      const float* __restrict__ g, const float* __restrict__ be,
                      float invcnt, float* __restrict__ a, float* __restrict__ c) {
  int i = threadIdx.x;
  if (i < 64) {
    float s = 0.f, q = 0.f;
    for (int p = 0; p < nparts; ++p) {
      s += parts[(size_t)p * 128 + i];
      q += parts[(size_t)p * 128 + 64 + i];
    }
    float mu  = s * invcnt;
    float var = q * invcnt - mu * mu;
    float sc  = g[i] * rsqrtf(var + 1e-5f);
    a[i] = sc; c[i] = be[i] - mu * sc;
  }
}

// ---------------------------------------------------------------------------
// Kernel: final BN3 + ReLU, float4-vectorized.
// ---------------------------------------------------------------------------
__global__ void k_apply(const float* __restrict__ hmax, const float* __restrict__ a3,
                        const float* __restrict__ c3, float* __restrict__ out) {
  int i = blockIdx.x * blockDim.x + threadIdx.x;   // float4 index
  const int total = ROWS_TOTAL * (CH / 4);
  for (; i < total; i += gridDim.x * blockDim.x) {
    float4 v = ((const float4*)hmax)[i];
    int c = (i & 15) << 2;
    float4 o;
    o.x = fmaxf(fmaf(v.x, a3[c],     c3[c]),     0.f);
    o.y = fmaxf(fmaf(v.y, a3[c + 1], c3[c + 1]), 0.f);
    o.z = fmaxf(fmaf(v.z, a3[c + 2], c3[c + 2]), 0.f);
    o.w = fmaxf(fmaf(v.w, a3[c + 3], c3[c + 3]), 0.f);
    ((float4*)out)[i] = o;
  }
}

// ---------------------------------------------------------------------------
extern "C" void kernel_launch(void* const* d_in, const int* in_sizes, int n_in,
                              void* d_out, int out_size, void* d_ws, size_t ws_size,
                              hipStream_t stream) {
  (void)in_sizes; (void)n_in; (void)out_size; (void)ws_size;
  const float* x   = (const float*)d_in[0];
  const float* W1  = (const float*)d_in[1];
  const float* b1  = (const float*)d_in[2];
  const float* g1  = (const float*)d_in[3];
  const float* be1 = (const float*)d_in[4];
  const float* W2  = (const float*)d_in[5];
  const float* b2  = (const float*)d_in[6];
  const float* g2  = (const float*)d_in[7];
  const float* be2 = (const float*)d_in[8];
  const float* gn  = (const float*)d_in[9];
  const float* ben = (const float*)d_in[10];
  float* out = (float*)d_out;

  // ---- workspace carve-up (256B aligned) ----
  uintptr_t base = (uintptr_t)d_ws;
  size_t off = 0;
  auto take = [&](size_t bytes) -> void* {
    void* p = (void*)(base + off);
    off = (off + bytes + 255) & ~(size_t)255;
    return p;
  };
  __bf16* xbf   = (__bf16*)take((size_t)ROWS_TOTAL * CIN * 2);       //   4 MB
  float*  sq    = (float*)take((size_t)ROWS_TOTAL * 4);              // 128 KB
  int*    knn   = (int*)take((size_t)ROWS_TOTAL * KNB * 4);          // 2.6 MB
  __bf16* W1t   = (__bf16*)take(2 * CIN * CH * 2);                   //  16 KB
  __bf16* W2t   = (__bf16*)take(CH * CH * 2);                        //   8 KB
  float*  part1 = (float*)take((size_t)1024 * 128 * 4);              // 512 KB
  float*  part2 = (float*)take((size_t)1024 * 128 * 4);              // 512 KB
  float*  part3 = (float*)take((size_t)128 * 128 * 4);               //  64 KB
  float*  coef  = (float*)take(6 * 64 * 4);                          // a1,c1,a2,c2,a3,c3
  float*  h1    = (float*)take((size_t)EROWS * CH * 4);              // 168 MB (reused as h2)
  float*  hmax  = (float*)take((size_t)ROWS_TOTAL * CH * 4);         // 8.4 MB
  float *a1c = coef, *c1c = coef + 64, *a2c = coef + 128, *c2c = coef + 192,
        *a3c = coef + 256, *c3c = coef + 320;

  const int knnLds = 16 * 4100 * 4;  // 262,400 B < 320 KB WGP LDS
  (void)hipFuncSetAttribute((const void*)k_knn,
                            hipFuncAttributeMaxDynamicSharedMemorySize, knnLds);

  // 1) prep: bf16 x + squared norms; bf16 transposed weights
  k_prep_x<<<ROWS_TOTAL / 8, 256, 0, stream>>>(x, xbf, sq);
  k_prep_w<<<48, 256, 0, stream>>>(W1, W2, W1t, W2t);

  // 2) fused distance GEMM (WMMA bf16) + top-20 in LDS (8 waves/WGP)
  k_knn<<<BB * (NN / 16), 256, knnLds, stream>>>(xbf, sq, knn);

  // 3) gather + edge MLP layer 1 (WMMA) + BN1 stats
  k_gemm1<<<1024, 128, 0, stream>>>(x, knn, W1t, b1, h1, part1);
  k_fin<<<1, 64, 0, stream>>>(part1, 1024, g1, be1, 1.0f / (float)EROWS, a1c, c1c);

  // 4) BN1+ReLU fused into layer 2 (WMMA, in-place) + BN2 stats
  k_gemm2<<<1024, 128, 0, stream>>>(h1, W2t, b2, a1c, c1c, part2);
  k_fin<<<1, 64, 0, stream>>>(part2, 1024, g2, be2, 1.0f / (float)EROWS, a2c, c2c);

  // 5) BN2+ReLU + max over K neighbors + BN3 stats
  k_maxk<<<128, 256, 0, stream>>>(h1, a2c, c2c, hmax, part3);
  k_fin<<<1, 64, 0, stream>>>(part3, 128, gn, ben, 1.0f / (float)ROWS_TOTAL, a3c, c3c);

  // 6) final BN3 + ReLU
  k_apply<<<2048, 256, 0, stream>>>(hmax, a3c, c3c, out);
}
